// LinearRNNEncoder_13915694039800
// MI455X (gfx1250) — compile-verified, matrix operand order checked by reading, not run on hardware
//
#include <hip/hip_runtime.h>
#include <hip/hip_bf16.h>
#include <math.h>

#define DEV __device__ __forceinline__

typedef __attribute__((ext_vector_type(16))) __bf16 v16bf;
typedef __attribute__((ext_vector_type(8)))  float  v8f;

constexpr int Bb = 8, Tt = 4096, Dd = 384, Ll = 8;
constexpr int Mtot = Bb * Tt;          // 32768 rows
constexpr int NC = 32, CT = Tt / NC;   // 32 chunks of 128 for the scan
constexpr float DECAY = 0.99f;

constexpr float decay_pow(int n){ double r = 1.0; for (int i = 0; i < n; ++i) r *= 0.99; return (float)r; }
constexpr float DEC_CT = decay_pow(CT);   // 0.99^128

// ---------- helpers ----------
DEV unsigned short f2bf(float f){
  unsigned int u = __float_as_uint(f);
  unsigned int r = u + 0x7fffu + ((u >> 16) & 1u);   // round-to-nearest-even
  return (unsigned short)(r >> 16);
}

DEV v16bf load_frag16(const unsigned short* p){
  union { uint4 u[2]; v16bf v; } f;
  f.u[0] = *reinterpret_cast<const uint4*>(p);        // K: base .. base+7
  f.u[1] = *reinterpret_cast<const uint4*>(p + 16);   // K: base+16 .. base+23
  return f.v;
}

DEV float sigm(float x){ return 1.0f / (1.0f + __expf(-x)); }

// ---------- bf16 GEMM with fused epilogues (C = A[M,K] * Wt[N,K]^T) ----------
// Wave tile 32(M) x 64(N): 2 A fragments, 8 accumulators, 8 WMMA per k-step.
// Ping-pong fragment buffers, unrolled x2: no register rotation copies.
// MODE 0: GATE  outF = sigmoid(acc+bias)*e0*mask        (Nc=384)
// MODE 1: FUSE  outF = outF + beta*e0 + (1-beta)*e1     (Nc=384)
// MODE 2: FFN1  outB = bf16(gelu(acc+bias))             (Nc=768)
// MODE 3: FFN2  outF += acc+bias                        (Nc=384)
template<int MODE, int K>
__global__ __launch_bounds__(256) void gemm_wmma(
    const unsigned short* __restrict__ A, const unsigned short* __restrict__ Wt,
    const float* __restrict__ bias, int Nc,
    const float* __restrict__ e0, const float* __restrict__ e1,
    const int* __restrict__ mask,
    float* __restrict__ outF, unsigned short* __restrict__ outB)
{
  const int lane = threadIdx.x & 31;
  const int wave = threadIdx.x >> 5;
  const int kg   = lane >> 4;       // K-half select (ISA 16-bit A layout)
  const int r16  = lane & 15;
  const int mBase = blockIdx.y * 128 + (wave & 3) * 32;
  const int nBase = blockIdx.x * 128 + (wave >> 2) * 64;

  const unsigned short* aRow0 = A  + (size_t)(mBase + r16) * K + 8 * kg;
  const unsigned short* aRow1 = aRow0 + (size_t)16 * K;
  const unsigned short* bRow  = Wt + (size_t)(nBase + r16) * K + 8 * kg;
  constexpr size_t bStep = (size_t)16 * K;

  v8f acc[2][4] = {};
  v16bf a[2][2], b[2][4];

  auto LD = [&](int buf, int kb){
    a[buf][0] = load_frag16(aRow0 + kb);
    a[buf][1] = load_frag16(aRow1 + kb);
    b[buf][0] = load_frag16(bRow + kb);
    b[buf][1] = load_frag16(bRow + bStep + kb);
    b[buf][2] = load_frag16(bRow + 2 * bStep + kb);
    b[buf][3] = load_frag16(bRow + 3 * bStep + kb);
  };
  auto MM = [&](int buf){
#pragma unroll
    for (int j = 0; j < 4; ++j){
      acc[0][j] = __builtin_amdgcn_wmma_f32_16x16x32_bf16(
          false, a[buf][0], false, b[buf][j], (short)0, acc[0][j], false, false);
      acc[1][j] = __builtin_amdgcn_wmma_f32_16x16x32_bf16(
          false, a[buf][1], false, b[buf][j], (short)0, acc[1][j], false, false);
    }
  };

  constexpr int steps = K / 32;        // 12 or 24 (even)
  LD(0, 0);
#pragma unroll
  for (int s = 0; s < steps; s += 2){
    int k1 = (s + 1) * 32; if (k1 > K - 32) k1 = K - 32;   // constant-folded
    LD(1, k1);
    MM(0);
    int k2 = (s + 2) * 32; if (k2 > K - 32) k2 = K - 32;   // clamped tail prefetch
    LD(0, k2);
    MM(1);
  }

#pragma unroll
  for (int h = 0; h < 2; ++h){
#pragma unroll
    for (int j = 0; j < 4; ++j){
      const int n = nBase + j * 16 + r16;
#pragma unroll
      for (int r = 0; r < 8; ++r){
        const int m = mBase + h * 16 + kg * 8 + r;   // C layout: lanes>=16 hold M=8..15
        const size_t g = (size_t)m * Nc + n;
        float v = acc[h][j][r] + bias[n];
        if (MODE == 0){
          float s = sigm(v);
          outF[g] = s * e0[g] * (float)mask[m];
        } else if (MODE == 1){
          float beta = sigm(v);
          outF[g] = outF[g] + beta * e0[g] + (1.0f - beta) * e1[g];
        } else if (MODE == 2){
          float gl = 0.5f * v * (1.0f + erff(v * 0.70710678118f));
          outB[g] = f2bf(gl);
        } else {
          outF[g] += v;
        }
      }
    }
  }
}

// ---------- weight convert + transpose: W[K,N] f32 -> Wt[N,K] bf16 ----------
__global__ void wconv_t(const float* __restrict__ W, unsigned short* __restrict__ Wt, int K, int N){
  int idx = blockIdx.x * 256 + threadIdx.x;
  if (idx >= K * N) return;
  int k = idx / N, n = idx - k * N;
  Wt[(size_t)n * K + k] = f2bf(W[idx]);
}

// ---------- embedding + LayerNorm ----------
__global__ __launch_bounds__(128) void embed_ln_k(
    const int* __restrict__ ids, const float* __restrict__ tok, const float* __restrict__ pos,
    const float* __restrict__ w, const float* __restrict__ bch, float* __restrict__ x)
{
  __shared__ float buf[Dd];
  __shared__ float red[128];
  const int m = blockIdx.x, tid = threadIdx.x;
  const int t = m & (Tt - 1);
  const int id = ids[m];
  float s = 0.f;
  for (int d = tid; d < Dd; d += 128){
    float v = tok[(size_t)id * Dd + d] + pos[(size_t)t * Dd + d];
    buf[d] = v; s += v;
  }
  red[tid] = s; __syncthreads();
  for (int o = 64; o > 0; o >>= 1){ if (tid < o) red[tid] += red[tid + o]; __syncthreads(); }
  float mu = red[0] * (1.0f / Dd);
  __syncthreads();
  float s2 = 0.f;
  for (int d = tid; d < Dd; d += 128){ float dv = buf[d] - mu; s2 += dv * dv; }
  red[tid] = s2; __syncthreads();
  for (int o = 64; o > 0; o >>= 1){ if (tid < o) red[tid] += red[tid + o]; __syncthreads(); }
  float rst = rsqrtf(red[0] * (1.0f / Dd) + 1e-5f);
  for (int d = tid; d < Dd; d += 128)
    x[(size_t)m * Dd + d] = (buf[d] - mu) * rst * w[d] + bch[d];
}

// ---------- generic 384-wide LayerNorm (optional f32 + bf16 outputs) ----------
__global__ __launch_bounds__(128) void ln384_k(
    const float* __restrict__ in, const float* __restrict__ w, const float* __restrict__ bch,
    float* __restrict__ outF, unsigned short* __restrict__ outB)
{
  __shared__ float buf[Dd];
  __shared__ float red[128];
  const int row = blockIdx.x, tid = threadIdx.x;
  const float* ip = in + (size_t)row * Dd;
  float s = 0.f;
  for (int d = tid; d < Dd; d += 128){ float v = ip[d]; buf[d] = v; s += v; }
  red[tid] = s; __syncthreads();
  for (int o = 64; o > 0; o >>= 1){ if (tid < o) red[tid] += red[tid + o]; __syncthreads(); }
  float mu = red[0] * (1.0f / Dd);
  __syncthreads();
  float s2 = 0.f;
  for (int d = tid; d < Dd; d += 128){ float dv = buf[d] - mu; s2 += dv * dv; }
  red[tid] = s2; __syncthreads();
  for (int o = 64; o > 0; o >>= 1){ if (tid < o) red[tid] += red[tid + o]; __syncthreads(); }
  float rst = rsqrtf(red[0] * (1.0f / Dd) + 1e-5f);
  for (int d = tid; d < Dd; d += 128){
    float o = (buf[d] - mu) * rst * w[d] + bch[d];
    if (outF) outF[(size_t)row * Dd + d] = o;
    if (outB) outB[(size_t)row * Dd + d] = f2bf(o);
  }
}

// ---------- blocked decayed scan: h_t = 0.99 h_{t-1} + v_t (in place) ----------
__global__ __launch_bounds__(384) void scan_local_k(float* __restrict__ hv, float* __restrict__ carry, int fwd){
  const int d = threadIdx.x, ch = blockIdx.x, b = blockIdx.y;
  float h = 0.f;
  for (int i = 0; i < CT; ++i){
    int p = ch * CT + i;
    int t = fwd ? p : (Tt - 1 - p);
    size_t idx = ((size_t)b * Tt + t) * Dd + d;
    h = fmaf(DECAY, h, hv[idx]);
    hv[idx] = h;
  }
  carry[((size_t)b * NC + ch) * Dd + d] = h;
}

__global__ __launch_bounds__(384) void scan_combine_k(const float* __restrict__ carry, float* __restrict__ pref){
  const int d = threadIdx.x, b = blockIdx.x;
  float c = 0.f;
  for (int ch = 0; ch < NC; ++ch){
    size_t idx = ((size_t)b * NC + ch) * Dd + d;
    pref[idx] = c;
    c = fmaf(DEC_CT, c, carry[idx]);
  }
}

__global__ __launch_bounds__(384) void scan_fix_k(
    float* __restrict__ hv, const float* __restrict__ pref,
    unsigned short* __restrict__ hcat, int fwd, int coff)
{
  const int d = threadIdx.x, ch = blockIdx.x, b = blockIdx.y;
  const float cin = pref[((size_t)b * NC + ch) * Dd + d];
  float fac = DECAY;
  for (int i = 0; i < CT; ++i){
    int p = ch * CT + i;
    int t = fwd ? p : (Tt - 1 - p);
    size_t idx = ((size_t)b * Tt + t) * Dd + d;
    float h = hv[idx] + fac * cin;
    hv[idx] = h;
    hcat[((size_t)b * Tt + t) * (2 * Dd) + coff + d] = f2bf(h);
    fac *= DECAY;
  }
}

// ---------- masked mean pool over T ----------
__global__ __launch_bounds__(384) void pool_mean_k(
    const float* __restrict__ xfin, const int* __restrict__ mask, float* __restrict__ pooled)
{
  const int b = blockIdx.x, d = threadIdx.x;
  float s = 0.f, ms = 0.f;
  for (int t = 0; t < Tt; ++t){
    float mf = (float)mask[b * Tt + t];
    s += xfin[((size_t)b * Tt + t) * Dd + d] * mf;
    ms += mf;
  }
  pooled[b * Dd + d] = s / fmaxf(ms, 1.0f);
}

// ---------- pooled @ poolW + b, LayerNorm -> embedding ----------
__global__ __launch_bounds__(384) void pool_proj_ln_k(
    const float* __restrict__ pooled, const float* __restrict__ Wp, const float* __restrict__ bp,
    const float* __restrict__ lw, const float* __restrict__ lb, float* __restrict__ out)
{
  __shared__ float buf[Dd];
  __shared__ float red[512];
  const int b = blockIdx.x, n = threadIdx.x;
  float acc = bp[n];
  for (int k = 0; k < Dd; ++k) acc += pooled[b * Dd + k] * Wp[(size_t)k * Dd + n];
  buf[n] = acc;
  red[n] = acc; if (n < 512 - Dd) red[Dd + n] = 0.f;
  __syncthreads();
  for (int o = 256; o > 0; o >>= 1){ if (n < o) red[n] += red[n + o]; __syncthreads(); }
  float mu = red[0] * (1.0f / Dd);
  __syncthreads();
  float dv = buf[n] - mu;
  red[n] = dv * dv; if (n < 512 - Dd) red[Dd + n] = 0.f;
  __syncthreads();
  for (int o = 256; o > 0; o >>= 1){ if (n < o) red[n] += red[n + o]; __syncthreads(); }
  float rst = rsqrtf(red[0] * (1.0f / Dd) + 1e-5f);
  out[b * Dd + n] = (buf[n] - mu) * rst * lw[n] + lb[n];
}

// ---------- host driver ----------
extern "C" void kernel_launch(void* const* d_in, const int* in_sizes, int n_in,
                              void* d_out, int out_size, void* d_ws, size_t ws_size,
                              hipStream_t stream)
{
  (void)in_sizes; (void)n_in; (void)out_size; (void)ws_size;
  const int*   ids   = (const int*)d_in[0];
  const int*   amask = (const int*)d_in[1];
  const float* tok   = (const float*)d_in[2];
  const float* pos   = (const float*)d_in[3];
  const float* enw   = (const float*)d_in[4];
  const float* enb   = (const float*)d_in[5];
  const float* n1w   = (const float*)d_in[6];
  const float* n1b   = (const float*)d_in[7];
  const float* n2w   = (const float*)d_in[8];
  const float* n2b   = (const float*)d_in[9];
  const float* gfW   = (const float*)d_in[10];
  const float* gfb   = (const float*)d_in[11];
  const float* gbW   = (const float*)d_in[12];
  const float* gbb   = (const float*)d_in[13];
  const float* fusW  = (const float*)d_in[14];
  const float* fusb  = (const float*)d_in[15];
  const float* f1W   = (const float*)d_in[16];
  const float* f1b   = (const float*)d_in[17];
  const float* f2W   = (const float*)d_in[18];
  const float* f2b   = (const float*)d_in[19];
  const float* fnw   = (const float*)d_in[20];
  const float* fnb   = (const float*)d_in[21];
  const float* pW    = (const float*)d_in[22];
  const float* pb    = (const float*)d_in[23];
  const float* pnw   = (const float*)d_in[24];
  const float* pnb   = (const float*)d_in[25];

  char* base = (char*)d_ws; size_t off = 0;
  auto carve = [&](size_t bytes)->char* {
    off = (off + 255) & ~(size_t)255;
    char* p = base + off; off += bytes; return p;
  };
  float* x    = (float*)carve((size_t)Mtot * Dd * 4);
  float* h32  = (float*)carve((size_t)Mtot * Dd * 4);
  float* hf   = (float*)carve((size_t)Mtot * Dd * 4);     // v_fwd, scanned in place
  float* hb   = (float*)carve((size_t)Mtot * Dd * 4);     // v_bwd, scanned in place
  unsigned short* hbf  = (unsigned short*)carve((size_t)Mtot * Dd * 2);
  unsigned short* h2bf = (unsigned short*)carve((size_t)Mtot * Dd * 2);
  unsigned short* hcat = (unsigned short*)carve((size_t)Mtot * 2 * Dd * 2);
  unsigned short* t1   = (unsigned short*)carve((size_t)Mtot * 2 * Dd * 2);
  float* carry  = (float*)carve((size_t)Bb * NC * Dd * 4);
  float* pref   = (float*)carve((size_t)Bb * NC * Dd * 4);
  float* pooled = (float*)carve((size_t)Bb * Dd * 4);
  unsigned short* wt_gf  = (unsigned short*)carve((size_t)Ll * Dd * Dd * 2);
  unsigned short* wt_gb  = (unsigned short*)carve((size_t)Ll * Dd * Dd * 2);
  unsigned short* wt_fus = (unsigned short*)carve((size_t)Ll * 2 * Dd * Dd * 2);
  unsigned short* wt_f1  = (unsigned short*)carve((size_t)Ll * 2 * Dd * Dd * 2);
  unsigned short* wt_f2  = (unsigned short*)carve((size_t)Ll * 2 * Dd * Dd * 2);

  float* outEmb = (float*)d_out;          // (B, DE)
  float* outX   = outEmb + Bb * Dd;       // (B, T, D) final-LN'd x

  // One-time (per launch) weight convert + transpose to bf16 N-major
  for (int l = 0; l < Ll; ++l){
    const int sq = Dd * Dd, rc = 2 * Dd * Dd;
    wconv_t<<<(sq + 255) / 256, 256, 0, stream>>>(gfW  + (size_t)l * sq, wt_gf  + (size_t)l * sq, Dd, Dd);
    wconv_t<<<(sq + 255) / 256, 256, 0, stream>>>(gbW  + (size_t)l * sq, wt_gb  + (size_t)l * sq, Dd, Dd);
    wconv_t<<<(rc + 255) / 256, 256, 0, stream>>>(fusW + (size_t)l * rc, wt_fus + (size_t)l * rc, 2 * Dd, Dd);
    wconv_t<<<(rc + 255) / 256, 256, 0, stream>>>(f1W  + (size_t)l * rc, wt_f1  + (size_t)l * rc, Dd, 2 * Dd);
    wconv_t<<<(rc + 255) / 256, 256, 0, stream>>>(f2W  + (size_t)l * rc, wt_f2  + (size_t)l * rc, 2 * Dd, Dd);
  }

  embed_ln_k<<<Mtot, 128, 0, stream>>>(ids, tok, pos, enw, enb, x);

  const dim3 g384(3, Mtot / 128), g768(6, Mtot / 128), gScan(NC, Bb);
  for (int l = 0; l < Ll; ++l){
    ln384_k<<<Mtot, 128, 0, stream>>>(x, n1w + l * Dd, n1b + l * Dd, h32, hbf);

    gemm_wmma<0, Dd><<<g384, 256, 0, stream>>>(hbf, wt_gf + (size_t)l * Dd * Dd, gfb + l * Dd,
                                               Dd, h32, nullptr, amask, hf, nullptr);
    gemm_wmma<0, Dd><<<g384, 256, 0, stream>>>(hbf, wt_gb + (size_t)l * Dd * Dd, gbb + l * Dd,
                                               Dd, h32, nullptr, amask, hb, nullptr);

    scan_local_k<<<gScan, 384, 0, stream>>>(hf, carry, 1);
    scan_combine_k<<<Bb, 384, 0, stream>>>(carry, pref);
    scan_fix_k<<<gScan, 384, 0, stream>>>(hf, pref, hcat, 1, 0);

    scan_local_k<<<gScan, 384, 0, stream>>>(hb, carry, 0);
    scan_combine_k<<<Bb, 384, 0, stream>>>(carry, pref);
    scan_fix_k<<<gScan, 384, 0, stream>>>(hb, pref, hcat, 0, Dd);

    gemm_wmma<1, 2 * Dd><<<g384, 256, 0, stream>>>(hcat, wt_fus + (size_t)l * 2 * Dd * Dd, fusb + l * Dd,
                                                   Dd, hf, hb, nullptr, x, nullptr);

    ln384_k<<<Mtot, 128, 0, stream>>>(x, n2w + l * Dd, n2b + l * Dd, nullptr, h2bf);

    gemm_wmma<2, Dd><<<g768, 256, 0, stream>>>(h2bf, wt_f1 + (size_t)l * 2 * Dd * Dd, f1b + l * 2 * Dd,
                                               2 * Dd, nullptr, nullptr, nullptr, nullptr, t1);
    gemm_wmma<3, 2 * Dd><<<g384, 256, 0, stream>>>(t1, wt_f2 + (size_t)l * 2 * Dd * Dd, f2b + l * Dd,
                                                   Dd, nullptr, nullptr, nullptr, x, nullptr);
  }

  ln384_k<<<Mtot, 128, 0, stream>>>(x, fnw, fnb, outX, nullptr);
  pool_mean_k<<<Bb, 384, 0, stream>>>(outX, amask, pooled);
  pool_proj_ln_k<<<Bb, 384, 0, stream>>>(pooled, pW, pb, pnw, pnb, outEmb);
}
